// FRMA_65343632441737
// MI455X (gfx1250) — matrix operand order repaired and checked by reading, not compiled, not run on hardware
//
#include <hip/hip_runtime.h>
#include <hip/hip_bf16.h>

// ---------------------------------------------------------------------------
// FRMA block for MI455X (gfx1250, wave32).
// All dense convolutions run as implicit-GEMM on v_wmma_f32_16x16x32_f16,
// register-blocked over output channels (MT tiles of 16 rows per wave) so the
// expensive im2col B-fragment build is amortized over MT WMMAs.
// Memory-bound stages (warp / layernorm / depthwise / 12x12 attention / geglu)
// run as plain VMEM/VALU kernels (they are HBM-limited, WMMA cannot help).
// ---------------------------------------------------------------------------

typedef __attribute__((ext_vector_type(16))) _Float16 v16h;
typedef __attribute__((ext_vector_type(8)))  float    v8f;

#define HH  192
#define WW  192
#define HWC 36864   // 192*192

// ---------------------------------------------------------------------------
// Generic implicit-GEMM conv:  out[m,n] = sum_k W[m,k] * im2col[k,n]
//   - two input segments (for channel-concatenated inputs), each fully strided
//     so 5-D (B,C,T,H,W) views and broadcasts (stride 0) work directly
//   - fused bias / leaky-relu(0.2) / residual-add epilogue
// Block = 8 waves covering 128 consecutive pixels. Each wave computes a
// (MT*16 Cout) x (16 pixel) tile: one B fragment, MT WMMAs per K-step.
// Weights staged in LDS pre-swizzled into WMMA A-fragment order.
// ---------------------------------------------------------------------------
struct ConvP {
  const float* inA; long asb, ast, acs; int atd, cinA;
  const float* inB; long bsb, bst, bcs; int btd, cinB;
  const float* wgt;                     // (Cout, Cin, KS, KS) f32
  const float* bias;                    // (Cout) or null
  const float* res;  long rsb, rst, rcs; int rtd;   // residual or null
  float*       out;  long osb, ost, ocs; int otd;
  int Cout, NI, act;                    // act: 1 = leaky_relu(0.2)
};

template<int KS, int MT>
__global__ __launch_bounds__(256) void conv_wmma_k(ConvP p) {
  constexpr int KK = KS * KS;
  const int wave  = threadIdx.x >> 5;
  const int lane  = threadIdx.x & 31;
  const int ln    = lane & 15;          // N (B frag) / M (A frag) index
  const int khalf = lane >> 4;          // lane group
  const int img   = blockIdx.z;
  const int m0    = blockIdx.y * (16 * MT);
  const int n0    = (blockIdx.x << 7) + (wave << 4);
  const int Cin   = p.cinA + p.cinB;
  const int Ktot  = Cin * KK;

  const long abase = (long)(img / p.atd) * p.asb + (long)(img % p.atd) * p.ast;
  const long bbase = p.inB ? (long)(img / p.btd) * p.bsb + (long)(img % p.btd) * p.bst : 0;

  __shared__ alignas(32) _Float16 Alds[MT * 512];  // MT*16 rows x 32 K, frag-swizzled

  v8f acc[MT];
  #pragma unroll
  for (int t = 0; t < MT; ++t) acc[t] = (v8f){0.f, 0.f, 0.f, 0.f, 0.f, 0.f, 0.f, 0.f};

  const int  n      = n0 + ln;
  const int  y      = n / WW;
  const int  x      = n - y * WW;
  const bool nvalid = (n < HWC);

  for (int k0 = 0; k0 < Ktot; k0 += 32) {
    __syncthreads();
    // ---- stage MT x (16x32) weight panel into LDS, swizzled to A-frag order:
    //      lane(M=m, group g) element j  <->  K = (j&7) + 8*g + 16*(j>>3)
    for (int idx = threadIdx.x; idx < MT * 512; idx += 256) {
      int mm = idx >> 5;                // 0 .. MT*16-1
      int kk = idx & 31;
      int kg = k0 + kk;
      float v = 0.f;
      int m = m0 + mm;
      if (m < p.Cout && kg < Ktot) {
        int c = kg / KK;
        int r = kg - c * KK;
        v = p.wgt[((long)m * Cin + c) * KK + r];
      }
      int g = (kk >> 3) & 1;
      int j = (kk & 7) | ((kk >> 4) << 3);
      Alds[(mm << 5) + (g << 4) + j] = (_Float16)v;
    }
    __syncthreads();

    // ---- build B fragment once (im2col, zero-pad), K = (j&7) + 8*lane_group + 16*(j>>3)
    v16h bfrag;
    #pragma unroll
    for (int j = 0; j < 16; ++j) {
      int kg = k0 + (j & 7) + (khalf << 3) + ((j >> 3) << 4);
      float v = 0.f;
      if (nvalid && kg < Ktot) {
        int c = kg / KK;
        int r = kg - c * KK;
        int yy = y, xx = x;
        if (KS == 3) { int ry = r / 3; yy += ry - 1; xx += (r - ry * 3) - 1; }
        if (yy >= 0 && yy < HH && xx >= 0 && xx < WW) {
          if (c < p.cinA)
            v = p.inA[abase + (long)c * p.acs + yy * WW + xx];
          else
            v = p.inB[bbase + (long)(c - p.cinA) * p.bcs + yy * WW + xx];
        }
      }
      bfrag[j] = (_Float16)v;
    }

    // ---- MT back-to-back WMMAs sharing the B fragment
    #pragma unroll
    for (int t = 0; t < MT; ++t) {
      v16h afrag = *(const v16h*)(Alds + (((t << 4) + ln) << 5) + (khalf << 4));
      acc[t] = __builtin_amdgcn_wmma_f32_16x16x32_f16(false, afrag, false, bfrag,
                                                      (short)0, acc[t], false, false);
    }
  }

  // ---- epilogue: C layout -> (VGPR v, lane) = (M = v + 8*(lane>>4), N = lane&15)
  const long obase = (long)(img / p.otd) * p.osb + (long)(img % p.otd) * p.ost;
  const long rbase = p.res ? (long)(img / p.rtd) * p.rsb + (long)(img % p.rtd) * p.rst : 0;
  #pragma unroll
  for (int t = 0; t < MT; ++t) {
    #pragma unroll
    for (int v = 0; v < 8; ++v) {
      int m  = m0 + (t << 4) + v + (khalf << 3);
      int nn = n0 + ln;
      if (m < p.Cout && nn < HWC) {
        float val = acc[t][v];
        if (p.bias) val += p.bias[m];
        if (p.act)  val = (val > 0.f) ? val : 0.2f * val;
        if (p.res)  val += p.res[rbase + (long)m * p.rcs + nn];
        p.out[obase + (long)m * p.ocs + nn] = val;
      }
    }
  }
}

// ---------------------------------------------------------------------------
// Depthwise 3x3 conv (groups == channels), contiguous (NI, C, H, W). HBM-bound.
// ---------------------------------------------------------------------------
__global__ void dwconv3_k(const float* __restrict__ in, const float* __restrict__ w,
                          float* __restrict__ out, int C, long total) {
  long idx = (long)blockIdx.x * 256 + threadIdx.x;
  if (idx >= total) return;
  int  n  = (int)(idx % HWC);
  long ic = idx / HWC;                  // i*C + c
  int  c  = (int)(ic % C);
  int  y  = n / WW, x = n - (n / WW) * WW;
  const float* ip = in + ic * HWC;
  const float* wp = w + (long)c * 9;
  float s = 0.f;
  #pragma unroll
  for (int dy = -1; dy <= 1; ++dy)
    #pragma unroll
    for (int dx = -1; dx <= 1; ++dx) {
      int yy = y + dy, xx = x + dx;
      if (yy >= 0 && yy < HH && xx >= 0 && xx < WW)
        s += ip[yy * WW + xx] * wp[(dy + 1) * 3 + (dx + 1)];
    }
  out[idx] = s;
}

// ---------------------------------------------------------------------------
// Channel layernorm over C=96 (+ optional fused add of F0 for MDTA input)
// ---------------------------------------------------------------------------
__global__ void layernorm_k(const float* __restrict__ xp0, const float* __restrict__ w,
                            const float* __restrict__ b, const float* __restrict__ addf,
                            float* __restrict__ out, long total) {
  long idx = (long)blockIdx.x * 256 + threadIdx.x;
  if (idx >= total) return;             // total = B*HW
  int n  = (int)(idx % HWC);
  int bb = (int)(idx / HWC);
  const float* xp = xp0 + (long)bb * 96 * HWC + n;
  float s = 0.f, s2 = 0.f;
  for (int c = 0; c < 96; ++c) { float v = xp[(long)c * HWC]; s += v; s2 += v * v; }
  float mu  = s * (1.f / 96.f);
  float var = s2 * (1.f / 96.f) - mu * mu;
  float inv = rsqrtf(var + 1e-5f);
  const float* fp = addf ? addf + (long)bb * 96 * HWC + n : (const float*)0;
  float* op = out + (long)bb * 96 * HWC + n;
  for (int c = 0; c < 96; ++c) {
    float v = (xp[(long)c * HWC] - mu) * inv * w[c] + b[c];
    if (fp) v += fp[(long)c * HWC];
    op[(long)c * HWC] = v;
  }
}

// ---------------------------------------------------------------------------
// Multiflow bilinear backwarp. Grid math collapses to sampling at (x+fx, y+fy).
// Fo/f in (B,C,T,H,W) layout; out in (B,96,T,H,W) == reshape (B, C*T, H, W).
// ---------------------------------------------------------------------------
__global__ void warp_k(const float* __restrict__ Fo, const float* __restrict__ f,
                       float* __restrict__ out, long total) {
  long idx = (long)blockIdx.x * 256 + threadIdx.x;
  if (idx >= total) return;             // total = B*8*3*HW
  int  n = (int)(idx % HWC);
  long r = idx / HWC;
  int  t  = (int)(r % 3); r /= 3;
  int  nf = (int)(r % 8);
  int  b  = (int)(r / 8);
  int  y  = n / WW, x = n - (n / WW) * WW;
  long fb = (((long)b * 24 + nf * 3) * 3 + t) * HWC + n;
  float fx  = f[fb];
  float fy  = f[fb + 3 * HWC];
  float fw  = f[fb + 6 * HWC];
  float wgt = 1.f / (1.f + __expf(-fw));
  float sx = (float)x + fx, sy = (float)y + fy;
  float flx = floorf(sx), fly = floorf(sy);
  int x0 = (int)flx, y0 = (int)fly;
  float wx1 = sx - flx, wy1 = sy - fly;
  float w00 = (1.f - wx1) * (1.f - wy1);
  float w10 = wx1 * (1.f - wy1);
  float w01 = (1.f - wx1) * wy1;
  float w11 = wx1 * wy1;
  bool vx0 = (x0 >= 0) & (x0 < WW), vx1 = (x0 + 1 >= 0) & (x0 + 1 < WW);
  bool vy0 = (y0 >= 0) & (y0 < HH), vy1 = (y0 + 1 >= 0) & (y0 + 1 < HH);
  int cx0 = min(max(x0, 0), WW - 1),     cx1 = min(max(x0 + 1, 0), WW - 1);
  int cy0 = min(max(y0, 0), HH - 1),     cy1 = min(max(y0 + 1, 0), HH - 1);
  float m00 = (vx0 && vy0) ? w00 : 0.f, m10 = (vx1 && vy0) ? w10 : 0.f;
  float m01 = (vx0 && vy1) ? w01 : 0.f, m11 = (vx1 && vy1) ? w11 : 0.f;
  for (int cd = 0; cd < 12; ++cd) {
    int c = nf * 12 + cd;
    const float* ip = Fo + (((long)b * 96 + c) * 3 + t) * HWC;
    float v = m00 * ip[cy0 * WW + cx0] + m10 * ip[cy0 * WW + cx1]
            + m01 * ip[cy1 * WW + cx0] + m11 * ip[cy1 * WW + cx1];
    out[(((long)b * 96 + c) * 3 + t) * HWC + n] = v * wgt;
  }
}

// ---------------------------------------------------------------------------
// Per-row L2 norms for q (channels 0..95) and k (96..191) of qkv (B,288,HW).
// One block per (b, row); LDS tree reduction.
// ---------------------------------------------------------------------------
__global__ __launch_bounds__(256) void l2norm_k(const float* __restrict__ qkv,
                                                float* __restrict__ norms) {
  int blk = blockIdx.x;                 // b*192 + r
  int b = blk / 192, r = blk - b * 192;
  const float* row = qkv + ((long)b * 288 + r) * HWC;
  float s = 0.f;
  for (int n = threadIdx.x; n < HWC; n += 256) { float v = row[n]; s += v * v; }
  __shared__ float red[256];
  red[threadIdx.x] = s;
  __syncthreads();
  for (int st = 128; st > 0; st >>= 1) {
    if (threadIdx.x < st) red[threadIdx.x] += red[threadIdx.x + st];
    __syncthreads();
  }
  if (threadIdx.x == 0) norms[blk] = fmaxf(sqrtf(red[0]), 1e-12f);
}

// ---------------------------------------------------------------------------
// attn[c,d] = (q_c . k_d) / (|q_c||k_d|) * temp[h].  One block per (b,head);
// 144 register accumulators per thread, LDS ds_add_f32 cross-thread reduce.
// ---------------------------------------------------------------------------
__global__ __launch_bounds__(256) void attn_qk_k(const float* __restrict__ qkv,
                                                 const float* __restrict__ norms,
                                                 const float* __restrict__ tempv,
                                                 float* __restrict__ attn) {
  int bh = blockIdx.x;
  int b = bh >> 3, h = bh & 7;
  const float* qb = qkv + ((long)b * 288 + h * 12) * HWC;
  const float* kb = qkv + ((long)b * 288 + 96 + h * 12) * HWC;
  float acc[144];
  #pragma unroll
  for (int i = 0; i < 144; ++i) acc[i] = 0.f;
  for (int n = threadIdx.x; n < HWC; n += 256) {
    float qv[12], kv[12];
    #pragma unroll
    for (int c = 0; c < 12; ++c) qv[c] = qb[(long)c * HWC + n];
    #pragma unroll
    for (int d = 0; d < 12; ++d) kv[d] = kb[(long)d * HWC + n];
    #pragma unroll
    for (int c = 0; c < 12; ++c)
      #pragma unroll
      for (int d = 0; d < 12; ++d)
        acc[c * 12 + d] = fmaf(qv[c], kv[d], acc[c * 12 + d]);
  }
  __shared__ float red[144];
  for (int i = threadIdx.x; i < 144; i += 256) red[i] = 0.f;
  __syncthreads();
  #pragma unroll
  for (int i = 0; i < 144; ++i) atomicAdd(&red[i], acc[i]);
  __syncthreads();
  for (int i = threadIdx.x; i < 144; i += 256) {
    int c = i / 12, d = i - c * 12;
    float qn = norms[b * 192 + h * 12 + c];
    float kn = norms[b * 192 + 96 + h * 12 + d];
    attn[(long)bh * 144 + i] = red[i] / (qn * kn) * tempv[h];
  }
}

__global__ void softmax12_k(float* __restrict__ attn, int rows) {
  int r = blockIdx.x * blockDim.x + threadIdx.x;
  if (r >= rows) return;
  float* row = attn + (long)r * 12;
  float mx = row[0];
  #pragma unroll
  for (int i = 1; i < 12; ++i) mx = fmaxf(mx, row[i]);
  float e[12], s = 0.f;
  #pragma unroll
  for (int i = 0; i < 12; ++i) { e[i] = __expf(row[i] - mx); s += e[i]; }
  float inv = 1.f / s;
  #pragma unroll
  for (int i = 0; i < 12; ++i) row[i] = e[i] * inv;
}

// out[h*12+c, n] = sum_d attn[c,d] * v[h*12+d, n]   (attn staged in LDS)
__global__ __launch_bounds__(256) void attn_v_k(const float* __restrict__ qkv,
                                                const float* __restrict__ attn,
                                                float* __restrict__ out) {
  int bh = blockIdx.y;
  int b = bh >> 3, h = bh & 7;
  __shared__ float a[144];
  for (int i = threadIdx.x; i < 144; i += 256) a[i] = attn[(long)bh * 144 + i];
  __syncthreads();
  int n = blockIdx.x * 256 + threadIdx.x;
  if (n >= HWC) return;
  const float* vb = qkv + ((long)b * 288 + 192 + h * 12) * HWC + n;
  float vv[12];
  #pragma unroll
  for (int d = 0; d < 12; ++d) vv[d] = vb[(long)d * HWC];
  float* ob = out + ((long)b * 96 + h * 12) * HWC + n;
  #pragma unroll
  for (int c = 0; c < 12; ++c) {
    float s = 0.f;
    #pragma unroll
    for (int d = 0; d < 12; ++d) s = fmaf(a[c * 12 + d], vv[d], s);
    ob[(long)c * HWC] = s;
  }
}

// exact gelu(y1) * y2, g = (B, 384, HW) -> out (B, 192, HW)
__global__ void geglu_k(const float* __restrict__ g, float* __restrict__ out, long total) {
  long idx = (long)blockIdx.x * 256 + threadIdx.x;
  if (idx >= total) return;
  int  n = (int)(idx % HWC);
  long r = idx / HWC;
  int  c = (int)(r % 192);
  int  b = (int)(r / 192);
  float y1 = g[((long)b * 384 + c) * HWC + n];
  float y2 = g[((long)b * 384 + 192 + c) * HWC + n];
  float ge = 0.5f * y1 * (1.f + erff(y1 * 0.70710678118654752f));
  out[idx] = ge * y2;
}

// ---------------------------------------------------------------------------
// Orchestration
// ---------------------------------------------------------------------------
extern "C" void kernel_launch(void* const* d_in, const int* in_sizes, int n_in,
                              void* d_out, int out_size, void* d_ws, size_t ws_size,
                              hipStream_t stream) {
  const long HW = 36864;
  const float* F     = (const float*)d_in[0];
  const float* F0    = (const float*)d_in[1];
  const float* dw0   = (const float*)d_in[2];
  const float* dw1   = (const float*)d_in[3];
  const float* dw2   = (const float*)d_in[4];
  const float* dw3   = (const float*)d_in[5];
  const float* rdbw  = (const float*)d_in[6];
  const float* rdbb  = (const float*)d_in[7];
  const float* cf1w  = (const float*)d_in[8];
  const float* cf2w  = (const float*)d_in[9];
  const float* fww   = (const float*)d_in[10];
  const float* n1w   = (const float*)d_in[11];
  const float* n1b   = (const float*)d_in[12];
  const float* tempv = (const float*)d_in[13];
  const float* qkvw  = (const float*)d_in[14];
  const float* qkvdw = (const float*)d_in[15];
  const float* projw = (const float*)d_in[16];
  const float* n2w   = (const float*)d_in[17];
  const float* n2b   = (const float*)d_in[18];
  const float* pinw  = (const float*)d_in[19];
  const float* ffndw = (const float*)d_in[20];
  const float* poutw = (const float*)d_in[21];

  // d_out = [Fo (B,96,3,H,W) | Fw (B,96,H,W) | f (B,24,3,H,W)]
  float* Fo   = (float*)d_out;
  float* Fw   = Fo + 21233664;
  float* fOut = Fw + 7077888;

  // workspace carving (regions reused across non-overlapping lifetimes)
  float* ws    = (float*)d_ws;
  float* bufA  = ws;                        // 28,311,552  : RDB growth (6x128xHW) / pin out (2x384xHW)
  float* bufB  = bufA + 28311552;           // 21,233,664  : warped (2x288xHW) / qkv (2x288xHW)
  float* bufC  = bufB + 21233664;           // 28,311,552  : qkv-dw (2x288xHW) / ffn-dw (2x384xHW)
  float* bufD  = bufC + 28311552;           //  7,077,888  : layernorm out (2x96xHW)
  float* bufE  = bufD + 7077888;            // 14,155,776  : cf1 tmp / attn-out / geglu out
  float* norms = bufE + 14155776;           // 384
  float* attnb = norms + 512;               // 2304

  // MT chosen so Cout is an exact multiple of 16*MT where possible:
  //   32 -> MT=2, 24 -> MT=2, 96/288 -> MT=3, 384 -> MT=4
  auto launch = [&](const ConvP& p, int KS, int MT) {
    dim3 g(288, (unsigned)((p.Cout + 16 * MT - 1) / (16 * MT)), (unsigned)p.NI);
    if (KS == 3) {
      if (MT == 2)      conv_wmma_k<3, 2><<<g, dim3(256), 0, stream>>>(p);
      else if (MT == 3) conv_wmma_k<3, 3><<<g, dim3(256), 0, stream>>>(p);
      else              conv_wmma_k<3, 4><<<g, dim3(256), 0, stream>>>(p);
    } else {
      if (MT == 2)      conv_wmma_k<1, 2><<<g, dim3(256), 0, stream>>>(p);
      else if (MT == 3) conv_wmma_k<1, 3><<<g, dim3(256), 0, stream>>>(p);
      else              conv_wmma_k<1, 4><<<g, dim3(256), 0, stream>>>(p);
    }
  };

  // ---- RDB: 4 dense 3x3 convs (leaky-relu) growing into bufA, then 1x1+res.
  // Input segment A = F viewed as 6 images of (96,HW) with 5-D strides.
  ConvP p{};
  p.inA = F;  p.asb = 96L * 3 * HW; p.ast = HW; p.acs = 3 * HW; p.atd = 3; p.cinA = 96;
  p.inB = nullptr; p.bsb = 0; p.bst = 0; p.bcs = HW; p.btd = 1; p.cinB = 0;
  p.bias = nullptr; p.res = nullptr; p.rsb = 0; p.rst = 0; p.rcs = 0; p.rtd = 1;
  p.osb = 128L * HW; p.ost = 0; p.ocs = HW; p.otd = 1;
  p.Cout = 32; p.NI = 6; p.act = 1;

  p.wgt = dw0; p.out = bufA;                          launch(p, 3, 2);
  p.inB = bufA; p.bsb = 128L * HW;
  p.cinB = 32;  p.wgt = dw1; p.out = bufA + 32 * HW;  launch(p, 3, 2);
  p.cinB = 64;  p.wgt = dw2; p.out = bufA + 64 * HW;  launch(p, 3, 2);
  p.cinB = 96;  p.wgt = dw3; p.out = bufA + 96 * HW;  launch(p, 3, 2);

  // rdb 1x1: Cin = [F(96) | dense(128)] = 224, bias, residual = F, out = Fo (5-D)
  p.cinB = 128; p.wgt = rdbw; p.bias = rdbb; p.act = 0;
  p.res = F; p.rsb = 96L * 3 * HW; p.rst = HW; p.rcs = 3 * HW; p.rtd = 3;
  p.out = Fo; p.osb = 96L * 3 * HW; p.ost = HW; p.ocs = 3 * HW; p.otd = 3;
  p.Cout = 96;                                        launch(p, 1, 3);

  // ---- flow: cf1 over concat([F0 broadcast over t, Fo]) -> lrelu -> cf2
  ConvP c1{};
  c1.inA = F0; c1.asb = 96L * HW; c1.ast = 0; c1.acs = HW; c1.atd = 3; c1.cinA = 96;
  c1.inB = Fo; c1.bsb = 96L * 3 * HW; c1.bst = HW; c1.bcs = 3 * HW; c1.btd = 3; c1.cinB = 96;
  c1.wgt = cf1w; c1.bias = nullptr; c1.res = nullptr; c1.rtd = 1;
  c1.out = bufE; c1.osb = 24L * HW; c1.ost = 0; c1.ocs = HW; c1.otd = 1;
  c1.Cout = 24; c1.NI = 6; c1.act = 1;                launch(c1, 3, 2);

  ConvP c2{};
  c2.inA = bufE; c2.asb = 24L * HW; c2.ast = 0; c2.acs = HW; c2.atd = 1; c2.cinA = 24;
  c2.inB = nullptr; c2.btd = 1; c2.cinB = 0;
  c2.wgt = cf2w; c2.bias = nullptr; c2.res = nullptr; c2.rtd = 1;
  c2.out = fOut; c2.osb = 24L * 3 * HW; c2.ost = HW; c2.ocs = 3 * HW; c2.otd = 3;
  c2.Cout = 24; c2.NI = 6; c2.act = 0;                launch(c2, 3, 2);

  // ---- multiflow backwarp: (B,96,3,HW) == w2d (B, 288, HW)
  {
    long total = 2L * 8 * 3 * HW;
    warp_k<<<dim3((unsigned)((total + 255) / 256)), dim3(256), 0, stream>>>(Fo, fOut, bufB, total);
  }

  // ---- convFw: 1x1, 288 -> 96, out = Fw (contiguous (B,96,HW))
  ConvP cw{};
  cw.inA = bufB; cw.asb = 288L * HW; cw.ast = 0; cw.acs = HW; cw.atd = 1; cw.cinA = 288;
  cw.inB = nullptr; cw.btd = 1; cw.cinB = 0;
  cw.wgt = fww; cw.bias = nullptr; cw.res = nullptr; cw.rtd = 1;
  cw.out = Fw; cw.osb = 96L * HW; cw.ost = 0; cw.ocs = HW; cw.otd = 1;
  cw.Cout = 96; cw.NI = 2; cw.act = 0;                launch(cw, 1, 3);

  // ---- 2x (MDTA attention + GEGLU FFN), residuals fused into conv epilogues
  for (int i = 0; i < 2; ++i) {
    // ln1(Fw) + F0  -> bufD
    layernorm_k<<<dim3(288), dim3(256), 0, stream>>>(Fw, n1w + i * 96, n1b + i * 96, F0, bufD, 2L * HW);

    // qkv 1x1: 96 -> 288 (WMMA) -> bufB
    ConvP q{};
    q.inA = bufD; q.asb = 96L * HW; q.ast = 0; q.acs = HW; q.atd = 1; q.cinA = 96;
    q.inB = nullptr; q.btd = 1; q.cinB = 0;
    q.wgt = qkvw + (long)i * 288 * 96; q.bias = nullptr; q.res = nullptr; q.rtd = 1;
    q.out = bufB; q.osb = 288L * HW; q.ost = 0; q.ocs = HW; q.otd = 1;
    q.Cout = 288; q.NI = 2; q.act = 0;                launch(q, 1, 3);

    // depthwise 3x3 on 288 ch -> bufC
    {
      long total = 2L * 288 * HW;
      dwconv3_k<<<dim3((unsigned)((total + 255) / 256)), dim3(256), 0, stream>>>(
          bufB, qkvdw + (long)i * 288 * 9, bufC, 288, total);
    }

    // channel attention (12x12 per head)
    l2norm_k<<<dim3(384), dim3(256), 0, stream>>>(bufC, norms);
    attn_qk_k<<<dim3(16), dim3(256), 0, stream>>>(bufC, norms, tempv + i * 8, attnb);
    softmax12_k<<<dim3(1), dim3(192), 0, stream>>>(attnb, 192);
    attn_v_k<<<dim3(144, 16), dim3(256), 0, stream>>>(bufC, attnb, bufE);

    // proj 1x1: 96 -> 96, residual += Fw, out = Fw
    ConvP pr{};
    pr.inA = bufE; pr.asb = 96L * HW; pr.ast = 0; pr.acs = HW; pr.atd = 1; pr.cinA = 96;
    pr.inB = nullptr; pr.btd = 1; pr.cinB = 0;
    pr.wgt = projw + (long)i * 96 * 96; pr.bias = nullptr;
    pr.res = Fw; pr.rsb = 96L * HW; pr.rst = 0; pr.rcs = HW; pr.rtd = 1;
    pr.out = Fw; pr.osb = 96L * HW; pr.ost = 0; pr.ocs = HW; pr.otd = 1;
    pr.Cout = 96; pr.NI = 2; pr.act = 0;              launch(pr, 1, 3);

    // ln2(Fw) -> bufD
    layernorm_k<<<dim3(288), dim3(256), 0, stream>>>(Fw, n2w + i * 96, n2b + i * 96, nullptr, bufD, 2L * HW);

    // pin 1x1: 96 -> 384 (WMMA) -> bufA
    ConvP pi{};
    pi.inA = bufD; pi.asb = 96L * HW; pi.ast = 0; pi.acs = HW; pi.atd = 1; pi.cinA = 96;
    pi.inB = nullptr; pi.btd = 1; pi.cinB = 0;
    pi.wgt = pinw + (long)i * 384 * 96; pi.bias = nullptr; pi.res = nullptr; pi.rtd = 1;
    pi.out = bufA; pi.osb = 384L * HW; pi.ost = 0; pi.ocs = HW; pi.otd = 1;
    pi.Cout = 384; pi.NI = 2; pi.act = 0;             launch(pi, 1, 4);

    // depthwise 3x3 on 384 ch -> bufC
    {
      long total = 2L * 384 * HW;
      dwconv3_k<<<dim3((unsigned)((total + 255) / 256)), dim3(256), 0, stream>>>(
          bufA, ffndw + (long)i * 384 * 9, bufC, 384, total);
    }

    // geglu: gelu(y1)*y2 -> bufE (B,192,HW)
    {
      long total = 2L * 192 * HW;
      geglu_k<<<dim3((unsigned)((total + 255) / 256)), dim3(256), 0, stream>>>(bufC, bufE, total);
    }

    // pout 1x1: 192 -> 96, residual += Fw, out = Fw
    ConvP po{};
    po.inA = bufE; po.asb = 192L * HW; po.ast = 0; po.acs = HW; po.atd = 1; po.cinA = 192;
    po.inB = nullptr; po.btd = 1; po.cinB = 0;
    po.wgt = poutw + (long)i * 96 * 192; po.bias = nullptr;
    po.res = Fw; po.rsb = 96L * HW; po.rst = 0; po.rcs = HW; po.rtd = 1;
    po.out = Fw; po.osb = 96L * HW; po.ost = 0; po.ocs = HW; po.otd = 1;
    po.Cout = 96; po.NI = 2; po.act = 0;              launch(po, 1, 3);
  }
}
